// NeighborhoodCrossAttention_31233002176666
// MI455X (gfx1250) — matrix operand order repaired and checked by reading, not compiled, not run on hardware
//
#include <hip/hip_runtime.h>
#include <math.h>

#define NA 32768
#define NB 32768
#define E_EDGES 262144
#define CA 512
#define CB 512
#define DMODEL 512
#define NHEAD 16
#define DH 32

// GEMM tiling
#define TM 128
#define TN 64
#define TK 32

typedef __attribute__((ext_vector_type(16))) __bf16 v16bf;
typedef __attribute__((ext_vector_type(8)))  float  v8f;

union Frag {
    unsigned u[8];
    v16bf    v;
};

// ---------------------------------------------------------------------------
// helpers: f32 <-> bf16 (round-to-nearest-even), split representation
// ---------------------------------------------------------------------------
__device__ __forceinline__ unsigned short f32_to_bf16_rne(float f) {
    unsigned u = __float_as_uint(f);
    u += 0x7FFFu + ((u >> 16) & 1u);
    return (unsigned short)(u >> 16);
}
__device__ __forceinline__ float bf16_to_f32(unsigned short h) {
    return __uint_as_float(((unsigned)h) << 16);
}

// ---------------------------------------------------------------------------
// Kernel 1: C[M,N] = X[M,K] * W[N,K]^T (+ R[M,N]) using split-bf16 WMMA.
//   x*y ~= hi_x*hi_y + hi_x*lo_y + lo_x*hi_y   (~fp32 accuracy, 3x bf16 rate)
// Requires M%128==0, N%64==0, K%32==0 (true for all launches here).
// Block: 256 threads (8 wave32). Tile: 128(M) x 64(N), K-step 32,
// double-buffered LDS with register prefetch of the next K-tile so global
// load latency hides behind the 12 WMMAs per wave per K-step.
// Wave (wm = wave&3, wn = wave>>2) owns a 32(M) x 32(N) sub-tile (2x2 D tiles).
// ---------------------------------------------------------------------------
__global__ __launch_bounds__(256) void gemm_xwt_splitbf16(
    const float* __restrict__ X, const float* __restrict__ W,
    const float* __restrict__ R, float* __restrict__ C,
    int M, int N, int K)
{
    __shared__ unsigned short sAhi[2][TM][34];
    __shared__ unsigned short sAlo[2][TM][34];
    __shared__ unsigned short sBhi[2][TN][34];
    __shared__ unsigned short sBlo[2][TN][34];

    const int tid  = threadIdx.x;
    const int lane = tid & 31;
    const int wave = tid >> 5;
    const int wm   = wave & 3;   // 0..3 -> M offset wm*32
    const int wn   = wave >> 2;  // 0..1 -> N offset wn*32
    const int m0   = blockIdx.y * TM;
    const int n0   = blockIdx.x * TN;

    v8f acc[2][2];
    acc[0][0] = (v8f){}; acc[0][1] = (v8f){};
    acc[1][0] = (v8f){}; acc[1][1] = (v8f){};

    float4 xa[4], xb[2];   // register prefetch staging (A: 128x32, B: 64x32)

    // ---- prologue: load K-tile 0 into registers, stage into LDS buffer 0
    #pragma unroll
    for (int i = 0; i < 4; ++i) {
        const int idx4 = tid + i * 256;
        xa[i] = *(const float4*)(X + (size_t)(m0 + (idx4 >> 3)) * K + ((idx4 & 7) << 2));
    }
    #pragma unroll
    for (int i = 0; i < 2; ++i) {
        const int idx4 = tid + i * 256;
        xb[i] = *(const float4*)(W + (size_t)(n0 + (idx4 >> 3)) * K + ((idx4 & 7) << 2));
    }
    {
        #pragma unroll
        for (int i = 0; i < 4; ++i) {
            const int idx4 = tid + i * 256;
            const int r = idx4 >> 3, c = (idx4 & 7) << 2;
            const float v4[4] = {xa[i].x, xa[i].y, xa[i].z, xa[i].w};
            #pragma unroll
            for (int j = 0; j < 4; ++j) {
                const unsigned short hi = f32_to_bf16_rne(v4[j]);
                sAhi[0][r][c + j] = hi;
                sAlo[0][r][c + j] = f32_to_bf16_rne(v4[j] - bf16_to_f32(hi));
            }
        }
        #pragma unroll
        for (int i = 0; i < 2; ++i) {
            const int idx4 = tid + i * 256;
            const int r = idx4 >> 3, c = (idx4 & 7) << 2;
            const float v4[4] = {xb[i].x, xb[i].y, xb[i].z, xb[i].w};
            #pragma unroll
            for (int j = 0; j < 4; ++j) {
                const unsigned short hi = f32_to_bf16_rne(v4[j]);
                sBhi[0][r][c + j] = hi;
                sBlo[0][r][c + j] = f32_to_bf16_rne(v4[j] - bf16_to_f32(hi));
            }
        }
    }
    __syncthreads();

    const int KT = K / TK;
    for (int t = 0; t < KT; ++t) {
        const int cur = t & 1;
        const bool have_next = (t + 1 < KT);
        const int kt_next = (t + 1) * TK;

        // ---- issue next-tile global loads first (latency hides under WMMA)
        if (have_next) {
            #pragma unroll
            for (int i = 0; i < 4; ++i) {
                const int idx4 = tid + i * 256;
                xa[i] = *(const float4*)(X + (size_t)(m0 + (idx4 >> 3)) * K + kt_next + ((idx4 & 7) << 2));
            }
            #pragma unroll
            for (int i = 0; i < 2; ++i) {
                const int idx4 = tid + i * 256;
                xb[i] = *(const float4*)(W + (size_t)(n0 + (idx4 >> 3)) * K + kt_next + ((idx4 & 7) << 2));
            }
        }

        // ---- A fragments (16x32 bf16): lane%16 = M row, lane/16 = K bank
        const int half = lane >> 4;
        Frag aHi[2], aLo[2];
        #pragma unroll
        for (int mi = 0; mi < 2; ++mi) {
            const int ar = wm * 32 + mi * 16 + (lane & 15);
            #pragma unroll
            for (int v = 0; v < 8; ++v) {
                // VGPR v, pair p: K = (v<4?0:16) + half*8 + (v&3)*2 + p
                const int kk = ((v & 4) << 2) + half * 8 + ((v & 3) << 1);
                aHi[mi].u[v] = *(const unsigned*)&sAhi[cur][ar][kk];
                aLo[mi].u[v] = *(const unsigned*)&sAlo[cur][ar][kk];
            }
        }
        // ---- B fragments (32x16 bf16): lane%16 = N col, lane/16 = K group
        const int kg = half << 4;
        Frag bHi[2], bLo[2];
        #pragma unroll
        for (int ni = 0; ni < 2; ++ni) {
            const int bc = wn * 32 + ni * 16 + (lane & 15);
            #pragma unroll
            for (int v = 0; v < 8; ++v) {
                const int kk = ((v & 4) << 2) + half * 8 + ((v & 3) << 1);        // K = (v<4?0:16) + half*8 + (v&3)*2 + p
                bHi[ni].u[v] = *(const unsigned*)&sBhi[cur][bc][kk];
                bLo[ni].u[v] = *(const unsigned*)&sBlo[cur][bc][kk];
            }
        }

        // ---- 12 WMMAs: 2x2 tiles x 3 split terms
        #pragma unroll
        for (int mi = 0; mi < 2; ++mi) {
            #pragma unroll
            for (int ni = 0; ni < 2; ++ni) {
                acc[mi][ni] = __builtin_amdgcn_wmma_f32_16x16x32_bf16(false, aHi[mi].v, false, bHi[ni].v, (short)0, acc[mi][ni], false, false);
                acc[mi][ni] = __builtin_amdgcn_wmma_f32_16x16x32_bf16(false, aHi[mi].v, false, bLo[ni].v, (short)0, acc[mi][ni], false, false);
                acc[mi][ni] = __builtin_amdgcn_wmma_f32_16x16x32_bf16(false, aLo[mi].v, false, bHi[ni].v, (short)0, acc[mi][ni], false, false);
            }
        }

        // ---- stage the prefetched tile into the other LDS buffer
        if (have_next) {
            const int nxt = cur ^ 1;
            #pragma unroll
            for (int i = 0; i < 4; ++i) {
                const int idx4 = tid + i * 256;
                const int r = idx4 >> 3, c = (idx4 & 7) << 2;
                const float v4[4] = {xa[i].x, xa[i].y, xa[i].z, xa[i].w};
                #pragma unroll
                for (int j = 0; j < 4; ++j) {
                    const unsigned short hi = f32_to_bf16_rne(v4[j]);
                    sAhi[nxt][r][c + j] = hi;
                    sAlo[nxt][r][c + j] = f32_to_bf16_rne(v4[j] - bf16_to_f32(hi));
                }
            }
            #pragma unroll
            for (int i = 0; i < 2; ++i) {
                const int idx4 = tid + i * 256;
                const int r = idx4 >> 3, c = (idx4 & 7) << 2;
                const float v4[4] = {xb[i].x, xb[i].y, xb[i].z, xb[i].w};
                #pragma unroll
                for (int j = 0; j < 4; ++j) {
                    const unsigned short hi = f32_to_bf16_rne(v4[j]);
                    sBhi[nxt][r][c + j] = hi;
                    sBlo[nxt][r][c + j] = f32_to_bf16_rne(v4[j] - bf16_to_f32(hi));
                }
            }
        }
        __syncthreads();
    }

    // ---- epilogue: C/D layout -> VGPR r holds M = r + 8*(lane/16), N = lane%16
    #pragma unroll
    for (int mi = 0; mi < 2; ++mi) {
        #pragma unroll
        for (int ni = 0; ni < 2; ++ni) {
            const int orow = m0 + wm * 32 + mi * 16 + (lane >> 4) * 8;
            const int ocol = n0 + wn * 32 + ni * 16 + (lane & 15);
            #pragma unroll
            for (int r = 0; r < 8; ++r) {
                const size_t o = (size_t)(orow + r) * N + ocol;
                float v = acc[mi][ni][r];
                if (R) v += R[o];
                C[o] = v;
            }
        }
    }
}

// ---------------------------------------------------------------------------
// Kernel 2: per-edge per-head logits: dot(Q[a_idx[e],h,:], K[b_idx[e],h,:])/sqrt(dh)
// one thread per (e,h); float4 loads; Q/K live in L2 (128 MB < 192 MB L2).
// ---------------------------------------------------------------------------
__global__ __launch_bounds__(256) void edge_logits_kernel(
    const float* __restrict__ Q, const float* __restrict__ Kmat,
    const int* __restrict__ a_idx, const int* __restrict__ b_idx,
    float* __restrict__ logits)
{
    const int idx = blockIdx.x * 256 + threadIdx.x;
    if (idx >= E_EDGES * NHEAD) return;
    const int e = idx >> 4;
    const int h = idx & 15;
    const float4* q = (const float4*)(Q    + (size_t)a_idx[e] * DMODEL + h * DH);
    const float4* k = (const float4*)(Kmat + (size_t)b_idx[e] * DMODEL + h * DH);
    float s = 0.f;
    #pragma unroll
    for (int i = 0; i < DH / 4; ++i) {
        float4 qv = q[i], kv = k[i];
        s += qv.x * kv.x + qv.y * kv.y + qv.z * kv.z + qv.w * kv.w;
    }
    logits[idx] = s * 0.17677669529663687f;   // 1/sqrt(32)
}

// ---------------------------------------------------------------------------
// Kernel 3: CSR row offsets via binary search (a_idx is sorted) — no atomics.
// ---------------------------------------------------------------------------
__global__ __launch_bounds__(256) void row_offsets_kernel(
    const int* __restrict__ a_idx, int* __restrict__ row_start)
{
    const int a = blockIdx.x * 256 + threadIdx.x;
    if (a > NA) return;
    int lo = 0, hi = E_EDGES;
    while (lo < hi) {
        const int mid = (lo + hi) >> 1;
        if (a_idx[mid] < a) lo = mid + 1; else hi = mid;
    }
    row_start[a] = lo;
}

// ---------------------------------------------------------------------------
// Kernel 4: segmented softmax + weighted-V scatter, fused. One wave per query.
// Lane -> (head = lane%16, dh half = lane/16); lane pair (l, l+16) redundantly
// recomputes the per-head max/denominator (deterministic, no LDS needed).
// ---------------------------------------------------------------------------
__global__ __launch_bounds__(256) void edge_softmax_av_kernel(
    const float* __restrict__ logits, const float* __restrict__ V,
    const int* __restrict__ b_idx, const int* __restrict__ row_start,
    float* __restrict__ out)
{
    const int wave = threadIdx.x >> 5;
    const int lane = threadIdx.x & 31;
    const int a    = blockIdx.x * 8 + wave;
    if (a >= NA) return;
    const int es = row_start[a];
    const int ee = row_start[a + 1];
    const int h  = lane & 15;
    const int d0 = (lane >> 4) << 4;   // 0 or 16

    float m = -INFINITY;
    for (int e = es; e < ee; ++e) m = fmaxf(m, logits[e * NHEAD + h]);
    float den = 0.f;
    for (int e = es; e < ee; ++e) den += __expf(logits[e * NHEAD + h] - m);
    const float inv = (den > 0.f) ? (1.0f / den) : 1.0f;

    float acc[16];
    #pragma unroll
    for (int j = 0; j < 16; ++j) acc[j] = 0.f;

    for (int e = es; e < ee; ++e) {
        const float w = __expf(logits[e * NHEAD + h] - m) * inv;
        const float4* vp = (const float4*)(V + (size_t)b_idx[e] * DMODEL + h * DH + d0);
        #pragma unroll
        for (int i = 0; i < 4; ++i) {
            const float4 vv = vp[i];
            acc[i * 4 + 0] += w * vv.x;
            acc[i * 4 + 1] += w * vv.y;
            acc[i * 4 + 2] += w * vv.z;
            acc[i * 4 + 3] += w * vv.w;
        }
    }

    float* op = out + (size_t)a * DMODEL + h * DH + d0;
    #pragma unroll
    for (int i = 0; i < 4; ++i) {
        float4 r;
        r.x = acc[i * 4 + 0]; r.y = acc[i * 4 + 1];
        r.z = acc[i * 4 + 2]; r.w = acc[i * 4 + 3];
        *(float4*)(op + i * 4) = r;
    }
}

// ---------------------------------------------------------------------------
// launch
// ---------------------------------------------------------------------------
extern "C" void kernel_launch(void* const* d_in, const int* in_sizes, int n_in,
                              void* d_out, int out_size, void* d_ws, size_t ws_size,
                              hipStream_t stream) {
    const float* Fa    = (const float*)d_in[0];   // (NA, CA)
    const float* Fb    = (const float*)d_in[1];   // (NB, CB)
    const float* Wq    = (const float*)d_in[2];   // (D, CA)
    const float* Wk    = (const float*)d_in[3];   // (D, CB)
    const float* Wv    = (const float*)d_in[4];   // (D, CB)
    const float* Wproj = (const float*)d_in[5];   // (CA, D)
    const int*   a_idx = (const int*)d_in[6];     // (E,)
    const int*   b_idx = (const int*)d_in[7];     // (E,)
    float*       outp  = (float*)d_out;           // (NA, CA)

    // workspace carve-up
    float* Q      = (float*)d_ws;                          // NA*D
    float* Kmat   = Q      + (size_t)NA * DMODEL;          // NB*D
    float* Vmat   = Kmat   + (size_t)NB * DMODEL;          // NB*D
    float* logits = Vmat   + (size_t)NB * DMODEL;          // E*H
    float* attn   = logits + (size_t)E_EDGES * NHEAD;      // NA*D
    int*   rows   = (int*)(attn + (size_t)NA * DMODEL);    // NA+1

    const dim3 blk(256);
    const dim3 ggrid(DMODEL / TN, NA / TM);   // (8, 256)

    // Q = Fa Wq^T ; K = Fb Wk^T ; V = Fb Wv^T
    gemm_xwt_splitbf16<<<ggrid, blk, 0, stream>>>(Fa, Wq, nullptr, Q,    NA, DMODEL, CA);
    gemm_xwt_splitbf16<<<ggrid, blk, 0, stream>>>(Fb, Wk, nullptr, Kmat, NB, DMODEL, CB);
    gemm_xwt_splitbf16<<<ggrid, blk, 0, stream>>>(Fb, Wv, nullptr, Vmat, NB, DMODEL, CB);

    // edge logits
    edge_logits_kernel<<<(E_EDGES * NHEAD) / 256, blk, 0, stream>>>(Q, Kmat, a_idx, b_idx, logits);

    // CSR offsets (a_idx sorted)
    row_offsets_kernel<<<(NA + 1 + 255) / 256, blk, 0, stream>>>(a_idx, rows);

    // fused segmented softmax + AV
    edge_softmax_av_kernel<<<NA / 8, blk, 0, stream>>>(logits, Vmat, b_idx, rows, attn);

    // out = Fa + attn Wproj^T
    gemm_xwt_splitbf16<<<ggrid, blk, 0, stream>>>(attn, Wproj, Fa, outp, NA, CA, DMODEL);
}
